// EncoderRNN_163208757264
// MI455X (gfx1250) — matrix operand order repaired and checked by reading, not compiled, not run on hardware
//
#include <hip/hip_runtime.h>

// EncoderRNN (Linear 64->256 + GRU H=256) for MI455X / gfx1250, wave32 WMMA.
//
//  - Fold Linear into GRU input gates: W_comb = W_ih @ W_lin  [768,64],
//    b_comb = W_ih @ b_lin + b_ih  -> gi computed on the fly (no 805MB buffer).
//  - Scan: 16 WGs x 16 batch rows, independent per batch row (no cross-WG sync).
//    512 threads = 16 waves; wave w owns h-columns [w*16, w*16+16) -> gate
//    tiles {w, 16+w, 32+w}; gate math + h-update fully in registers (h
//    persistent in regs across all 1024 steps).
//  - Weights pre-swizzled to wave32 B-fragment layout: 2x global_load_b128 per
//    fragment, L2-resident (480KB total). A laundered 32-bit offset (always 0,
//    opaqued via empty asm) makes fragment addresses non-invariant so LICM
//    cannot hoist 240 VGPRs of fragments out of the loop (previously caused
//    scratch spills), while keeping global (not flat) addressing.
//  - h shared state: 8KB f16 LDS buffer already in A-fragment layout;
//    update scatters ds_store_b16, GEMM reads 2x ds_load_b128 per K-step.

#define IN_  64
#define H_   256
#define B_   256
#define T_   1024
#define G3_  768

typedef _Float16 v16h __attribute__((ext_vector_type(16)));
typedef float    v8f  __attribute__((ext_vector_type(8)));

union FragA { v16h v; _Float16 h[16]; uint4 q[2]; };

static __device__ __forceinline__ v8f splat8(float f) {
  v8f r = {f, f, f, f, f, f, f, f};
  return r;
}

static __device__ __forceinline__ v8f wmma16(const FragA& a, const FragA& b, v8f c) {
  // D = A(16x32 f16) * B(32x16 f16) + C(16x16 f32)
  return __builtin_amdgcn_wmma_f32_16x16x32_f16(
      /*neg_a=*/false, a.v, /*neg_b=*/false, b.v,
      /*c_mod=*/(short)0, c, /*reuse_a=*/false, /*reuse_b=*/false);
}

// Swizzled fragment: 512 f16 = 1KB per (tile,kk); lane's 16 values contiguous (32B).
static __device__ __forceinline__ void load_frag(FragA& b, const _Float16* base,
                                                 int frag, int lane) {
  const uint4* p = reinterpret_cast<const uint4*>(base) + (size_t)frag * 64 + lane * 2;
  b.q[0] = p[0];
  b.q[1] = p[1];
}

// ---------------- prep kernels ----------------

// W_comb[g,i] = sum_k W_ih[g,k] * W_lin[k,i], written f16-swizzled (KK=2).
__global__ void prep_comb(const float* __restrict__ W_lin,
                          const float* __restrict__ W_ih,
                          _Float16* __restrict__ wc_sw) {
  int idx = blockIdx.x * blockDim.x + threadIdx.x;
  if (idx >= G3_ * IN_) return;
  int g = idx >> 6;
  int i = idx & 63;
  float acc = 0.f;
  for (int k = 0; k < H_; ++k) acc += W_ih[g * H_ + k] * W_lin[k * IN_ + i];
  int tile = g >> 4, kk = i >> 5;
  int lane = (g & 15) | (((i >> 3) & 1) << 4);
  int v = ((i >> 4) & 1) * 4 + ((i & 7) >> 1), par = i & 1;
  wc_sw[(size_t)((tile * 2 + kk) << 9) + lane * 16 + v * 2 + par] = (_Float16)acc;
}

// W_hh converted to f16 and swizzled (KK=8).
__global__ void prep_whh(const float* __restrict__ W_hh, _Float16* __restrict__ wh_sw) {
  int idx = blockIdx.x * blockDim.x + threadIdx.x;
  if (idx >= G3_ * H_) return;
  int g = idx >> 8;
  int k = idx & 255;
  float w = W_hh[idx];
  int tile = g >> 4, kk = k >> 5;
  int lane = (g & 15) | (((k >> 3) & 1) << 4);
  int v = ((k >> 4) & 1) * 4 + ((k & 7) >> 1), par = k & 1;
  wh_sw[(size_t)((tile * 8 + kk) << 9) + lane * 16 + v * 2 + par] = (_Float16)w;
}

// b_comb[g] = W_ih[g,:] . b_lin + b_ih[g]
__global__ void prep_bc(const float* __restrict__ W_ih, const float* __restrict__ b_lin,
                        const float* __restrict__ b_ih, float* __restrict__ bcomb) {
  int g = blockIdx.x * blockDim.x + threadIdx.x;
  if (g >= G3_) return;
  float acc = b_ih[g];
  for (int k = 0; k < H_; ++k) acc += W_ih[g * H_ + k] * b_lin[k];
  bcomb[g] = acc;
}

// ---------------- GRU scan ----------------

__global__ __launch_bounds__(512, 1)
void gru_scan(const float* __restrict__ x,       // [B,T,64]
              const float* __restrict__ h0,      // [1,B,256]
              const float* __restrict__ b_hh,    // [768]
              const _Float16* __restrict__ wc,   // swizzled W_comb [48*2 frags]
              const _Float16* __restrict__ wh,   // swizzled W_hh   [48*8 frags]
              const float* __restrict__ bc,      // [768]
              float* __restrict__ out)           // [B,T,H] ++ [B,H]
{
  // h state shared across waves, stored directly in WMMA A-fragment layout.
  __shared__ _Float16 hfrag[8 * 512];  // 8 K-steps x 1KB fragments

  const int tid   = threadIdx.x;
  const int wave  = tid >> 5;    // 0..15
  const int lane  = tid & 31;
  const int lhalf = lane >> 4;
  const int lm    = lane & 15;
  const int b0    = blockIdx.x << 4;

  // Gate tiles owned by this wave (same 16 h-columns for r/z/n).
  const int tR = wave;        // gates [0,256)
  const int tZ = 16 + wave;   // gates [256,512)
  const int tN = 32 + wave;   // gates [512,768)
  const int jj = wave * 16 + lm;  // owned h column

  // Biases folded into accumulator init (column jj, same for all 8 rows).
  const float bRv  = bc[jj] + b_hh[jj];
  const float bZv  = bc[256 + jj] + b_hh[256 + jj];
  const float bINv = bc[512 + jj];
  const float bHNv = b_hh[512 + jj];

  // Scatter position of element (row M, col jj) inside hfrag (A layout):
  // lane' = M + ((jj>>3)&1)*16 ; reg v = (Kl>>4)*4 + ((Kl&7)>>1) ; parity = jj&1.
  const int Kl    = jj & 31;
  const int halfK = (Kl >> 3) & 1;
  const int vK    = (Kl >> 4) * 4 + ((Kl & 7) >> 1);
  const int fbase = (jj >> 5) * 512 + vK * 2 + (jj & 1);

  // h persistent in registers; build initial fragment buffer.
  float hh[8];
#pragma unroll
  for (int r = 0; r < 8; ++r) {
    int row = r + lhalf * 8;
    hh[r] = h0[(size_t)(b0 + row) * H_ + jj];
    hfrag[fbase + (row + halfK * 16) * 16] = (_Float16)hh[r];
  }
  __syncthreads();

  const float* xrow = x + (size_t)(b0 + lm) * T_ * IN_;  // A-matrix row M = lm

#pragma unroll 1
  for (int t = 0; t < T_; ++t) {
    // Laundered zero offset: blocks LICM from hoisting the 30 weight fragments
    // (240 VGPRs) out of the t-loop, but keeps the base pointers traceable to
    // kernargs so the loads stay on the GLOBAL path (not flat): weight
    // fragments re-read from L2 each step by design.
    unsigned off = 0;
    asm volatile("" : "+s"(off));
    const _Float16* wcp = wc + off;
    const _Float16* whp = wh + off;

    v8f aR  = splat8(bRv);
    v8f aZ  = splat8(bZv);
    v8f aIN = splat8(bINv);
    v8f aHN = splat8(bHNv);

    const float* xp = xrow + (size_t)t * IN_;
    if (t + 1 < T_) __builtin_prefetch(xp + IN_, 0, 3);  // global_prefetch next X row

    // ---- input contribution: gi = X_t * W_comb^T (K = 64, 2 sub-steps) ----
#pragma unroll
    for (int kk = 0; kk < 2; ++kk) {
      FragA a;
#pragma unroll
      for (int v = 0; v < 8; ++v) {
        int K = kk * 32 + ((v >> 2) << 4) + (lhalf << 3) + ((v & 3) << 1);
        float2 xv = *(const float2*)(xp + K);
        a.h[2 * v]     = (_Float16)xv.x;
        a.h[2 * v + 1] = (_Float16)xv.y;
      }
      FragA bb;
      load_frag(bb, wcp, tR * 2 + kk, lane); aR  = wmma16(a, bb, aR);
      load_frag(bb, wcp, tZ * 2 + kk, lane); aZ  = wmma16(a, bb, aZ);
      load_frag(bb, wcp, tN * 2 + kk, lane); aIN = wmma16(a, bb, aIN);
    }

    // ---- recurrent contribution: gh = h * W_hh^T (K = 256, 8 sub-steps) ----
#pragma unroll
    for (int kk = 0; kk < 8; ++kk) {
      FragA a;  // h already stored in fragment layout: 2x ds_load_b128
      const uint4* hp = reinterpret_cast<const uint4*>(hfrag + kk * 512) + lane * 2;
      a.q[0] = hp[0];
      a.q[1] = hp[1];
      FragA bb;
      load_frag(bb, whp, tR * 8 + kk, lane); aR  = wmma16(a, bb, aR);
      load_frag(bb, whp, tZ * 8 + kk, lane); aZ  = wmma16(a, bb, aZ);
      load_frag(bb, whp, tN * 8 + kk, lane); aHN = wmma16(a, bb, aHN);
    }

    __syncthreads();  // all waves done reading hfrag

    // ---- gate nonlinearity + h update, fully in registers ----
#pragma unroll
    for (int r = 0; r < 8; ++r) {
      int row = r + lhalf * 8;  // C/D layout: lanes 16-31 hold M = r+8
      float rr = 1.f / (1.f + __expf(-aR[r]));
      float zz = 1.f / (1.f + __expf(-aZ[r]));
      float nn = tanhf(aIN[r] + rr * aHN[r]);
      float hnew = (1.f - zz) * nn + zz * hh[r];
      hh[r] = hnew;
      hfrag[fbase + (row + halfK * 16) * 16] = (_Float16)hnew;
      // streaming output: don't pollute L2 (weights live there)
      __builtin_nontemporal_store(
          hnew, out + (size_t)(b0 + row) * T_ * H_ + (size_t)t * H_ + jj);
    }
    __syncthreads();  // hfrag updates visible before next step's reads
  }

  // final hidden state [1,B,H]
#pragma unroll
  for (int r = 0; r < 8; ++r) {
    int row = r + lhalf * 8;
    out[(size_t)B_ * T_ * H_ + (size_t)(b0 + row) * H_ + jj] = hh[r];
  }
}

// ---------------- launch ----------------

extern "C" void kernel_launch(void* const* d_in, const int* in_sizes, int n_in,
                              void* d_out, int out_size, void* d_ws, size_t ws_size,
                              hipStream_t stream) {
  const float* inp    = (const float*)d_in[0];  // [B,T,64]
  const float* hidden = (const float*)d_in[1];  // [1,B,256]
  const float* W_lin  = (const float*)d_in[2];  // [256,64]
  const float* b_lin  = (const float*)d_in[3];  // [256]
  const float* W_ih   = (const float*)d_in[4];  // [768,256]
  const float* b_ih   = (const float*)d_in[5];  // [768]
  const float* W_hh   = (const float*)d_in[6];  // [768,256]
  const float* b_hh   = (const float*)d_in[7];  // [768]
  float* out = (float*)d_out;

  char* ws = (char*)d_ws;
  _Float16* wc_sw = (_Float16*)ws;                       //  98304 B
  _Float16* wh_sw = (_Float16*)(ws + 98304);             // 393216 B
  float*    bcomb = (float*)(ws + 98304 + 393216);       //   3072 B

  prep_comb<<<(G3_ * IN_ + 255) / 256, 256, 0, stream>>>(W_lin, W_ih, wc_sw);
  prep_whh<<<(G3_ * H_ + 255) / 256, 256, 0, stream>>>(W_hh, wh_sw);
  prep_bc<<<(G3_ + 255) / 256, 256, 0, stream>>>(W_ih, b_lin, b_ih, bcomb);
  gru_scan<<<B_ / 16, 512, 0, stream>>>(inp, hidden, b_hh, wc_sw, wh_sw, bcomb, out);
}
